// RNN_MultiRegional_SAC_85143431676477
// MI455X (gfx1250) — matrix operand order repaired and checked by reading, not compiled, not run on hardware
//
#include <hip/hip_runtime.h>

// ---------------------------------------------------------------------------
// RNN_MultiRegional scan on gfx1250 (MI455X): persistent WMMA kernel.
//   h_{t+1} = relu(0.9*h + 0.1*(h @ W_rec^T + drive_t)),  h:[32,3072], T=512
// W_rec (f16, 19MB) stays L2-resident; block-sparsity (5/9 blocks) exploited
// by per-column-block K-segments. Grid-wide barrier between steps.
// Round-4 changes: 8-deep software pipeline in the GEMM segment loop so the
// load->wmma wait distance spans a full 8-WMMA iteration (covers L2 latency
// with 1 wave/SIMD; partial loadcnt waits instead of s_wait_loadcnt 0x0).
// ---------------------------------------------------------------------------

typedef __attribute__((ext_vector_type(16))) _Float16 v16h;
typedef __attribute__((ext_vector_type(8)))  _Float16 v8h;
typedef __attribute__((ext_vector_type(8)))  float    v8f;

#define HH     1024
#define H3     3072
#define BB     32
#define TT     512
#define N_INH  307               // int(0.3*1024)
#define HS     ((long)BB * H3)   // elements in one h buffer (98304)
#define NBLOCKS_SCAN 96          // 3072/32 column-slices
#define DEPTH  8                 // k-tiles in flight in the GEMM pipeline
#define OUT_MEAN_OFF 0
#define OUT_STD_OFF  16384
#define OUT_HN_OFF   32768
#define OUT_RNN_OFF  131072

__device__ __forceinline__ float ht_clip(float w) {
    return fminf(fmaxf(w, 1e-10f), 1.0f);   // F.hardtanh(min=1e-10, max=1)
}

// ---- build f16 W_rec [3072 x 3072], row-major (row n = output, col k) ------
__global__ void build_wrec_kernel(const float* __restrict__ w_str2thal,
                                  const float* __restrict__ w_m12m1,
                                  const float* __restrict__ w_m12str,
                                  const float* __restrict__ w_thal2m1,
                                  const float* __restrict__ str2str_fixed,
                                  _Float16* __restrict__ W16) {
    long idx = (long)blockIdx.x * 256 + threadIdx.x;
    if (idx >= (long)H3 * H3) return;
    int n = (int)(idx / H3), k = (int)(idx % H3);
    int nb = n >> 10, kb = k >> 10;
    int nn = n & (HH - 1), kk = k & (HH - 1);
    float v = 0.0f;
    if (nb == 0 && kb == 0) {
        // str2str_mask == 0 -> only -fixed survives
        v = -str2str_fixed[(long)nn * HH + kk];
    } else if (nb == 0 && kb == 2) {
        v = (kk < HH - N_INH) ? ht_clip(w_m12str[(long)nn * HH + kk]) : 0.0f;
    } else if (nb == 1 && kb == 0) {
        v = ht_clip(w_str2thal[(long)nn * HH + kk]) * ((kk < HH / 2) ? 1.0f : -1.0f);
    } else if (nb == 2 && kb == 1) {
        v = ht_clip(w_thal2m1[(long)nn * HH + kk]);
    } else if (nb == 2 && kb == 2) {
        v = ht_clip(w_m12m1[(long)nn * HH + kk]) * ((kk < HH - N_INH) ? 1.0f : -1.0f);
    }
    W16[idx] = (_Float16)v;
}

// ---- init h state (f32 + f16 shadow) and sync counters ---------------------
__global__ void init_state_kernel(const float* __restrict__ hn,
                                  float* __restrict__ h32,
                                  _Float16* __restrict__ h16,
                                  unsigned* __restrict__ sync) {
    int i = blockIdx.x * 256 + threadIdx.x;
    if (i < BB * H3) {
        float v = hn[i];
        h32[i] = v;
        h16[i] = (_Float16)v;
    }
    if (i == 0) { sync[0] = 0u; sync[1] = 0u; }
}

// ---- grid-wide barrier (generation counter) --------------------------------
__device__ __forceinline__ void grid_barrier(unsigned* cnt, unsigned* gen,
                                             unsigned nblocks) {
    __syncthreads();
    if (threadIdx.x == 0) {
        __threadfence();
        unsigned g = atomicAdd(gen, 0u);
        if (atomicAdd(cnt, 1u) == nblocks - 1u) {
            atomicExch(cnt, 0u);
            __threadfence();
            atomicAdd(gen, 1u);
        } else {
            while (atomicAdd(gen, 0u) == g) __builtin_amdgcn_s_sleep(2);
        }
        __threadfence();
    }
    __syncthreads();
}

union U16 { v16h v; v8h h[2]; };

#define WMMA(Acc, Au, Bu)                                              \
    (Acc) = __builtin_amdgcn_wmma_f32_16x16x32_f16(                    \
        false, (Au).v, false, (Bu).v, (short)0, (Acc), false, false)

#define LOAD_TILE(Ad, Bd, off_) do {                                   \
        (Ad).h[0] = *(const v8h*)(a + (off_));                         \
        (Ad).h[1] = *(const v8h*)(a + (off_) + 8);                     \
        (Bd).v    = *(const v16h*)(b + (off_));                        \
    } while (0)

// One contiguous K segment; DEPTH k-tiles of loads in flight past the WMMAs.
// klen is a multiple of DEPTH*32 (1024 or 2048).
__device__ __forceinline__ v8f gemm_segment(v8f acc,
                                            const _Float16* __restrict__ Arow,
                                            const _Float16* __restrict__ Brow,
                                            int kbeg, int klen) {
    const _Float16* a = Arow + kbeg;
    const _Float16* b = Brow + kbeg;
    U16 A[DEPTH], Bv[DEPTH];
    #pragma unroll
    for (int d = 0; d < DEPTH; ++d)
        LOAD_TILE(A[d], Bv[d], d * 32);
    for (int k = 0; k < klen - DEPTH * 32; k += DEPTH * 32) {
        #pragma unroll
        for (int d = 0; d < DEPTH; ++d) {
            WMMA(acc, A[d], Bv[d]);
            LOAD_TILE(A[d], Bv[d], k + (DEPTH + d) * 32);
        }
    }
    #pragma unroll
    for (int d = 0; d < DEPTH; ++d)
        WMMA(acc, A[d], Bv[d]);
    return acc;
}

// ---- persistent scan kernel: 96 blocks x 128 threads (4 waves) -------------
// wave w: m-tile = 16*(w&1), n-tile = blockIdx.x*32 + 16*(w>>1)
__global__ __launch_bounds__(128, 1)
void rnn_scan_kernel(const _Float16* __restrict__ W16,
                     const float*    __restrict__ inp,    // [32,512,4]
                     const float*    __restrict__ inpw,   // [4,3072]
                     float*    __restrict__ h32,          // 2 * HS floats
                     _Float16* __restrict__ h16,          // 2 * HS halves
                     float* __restrict__ out,             // full d_out base
                     unsigned* __restrict__ sync) {
    const int wv    = threadIdx.x >> 5;
    const int lane  = threadIdx.x & 31;
    const int laneL = lane & 15;
    const int laneH = lane >> 4;

    const int m0    = (wv & 1) * 16;
    const int ncol0 = blockIdx.x * 32 + (wv >> 1) * 16;
    const int nb    = ncol0 >> 10;   // uniform across the block (32-col slices)

    // nonzero K segments of W_rec for this output column block
    int s0, l0, s1, l1;
    if (nb == 0)      { s0 = 0;    l0 = HH;     s1 = 2 * HH; l1 = HH; }
    else if (nb == 1) { s0 = 0;    l0 = HH;     s1 = 0;      l1 = 0;  }
    else              { s0 = HH;   l0 = 2 * HH; s1 = 0;      l1 = 0;  }

    const int n = ncol0 + laneL;            // this lane's output column
    // B operand: row-major W, lane covers row n, 16 consecutive K per laneH half
    const _Float16* Brow = W16 + (long)n * H3 + laneH * 16;
    // per-lane input-weight column (drive computed on the fly)
    const float iw0 = inpw[0 * H3 + n];
    const float iw1 = inpw[1 * H3 + n];
    const float iw2 = inpw[2 * H3 + n];
    const float iw3 = inpw[3 * H3 + n];

    const int am = m0 + laneL;              // A-operand row for this lane
    const float4* inp4 = reinterpret_cast<const float4*>(inp);

    float* out_rnn = out + OUT_RNN_OFF;
    float* out_hn  = out + OUT_HN_OFF;

    for (int t = 0; t < TT; ++t) {
        const long curo = (long)(t & 1) * HS;
        const long nxto = (long)((t & 1) ^ 1) * HS;
        const _Float16* __restrict__ Arow = h16 + curo + (long)am * H3 + laneH * 16;
        const float*    __restrict__ hc32 = h32 + curo;
        float*          __restrict__ hx32 = h32 + nxto;
        _Float16*       __restrict__ hx16 = h16 + nxto;

        v8f acc = {};
        acc = gemm_segment(acc, Arow, Brow, s0, l0);
        if (l1)                                   // block-uniform branch
            acc = gemm_segment(acc, Arow, Brow, s1, l1);

        // pointwise update: relu(0.9*h + 0.1*(acc + drive)) for 8 outputs/lane
        // C/D layout: vgpr r -> M = m0 + r + 8*laneH, N = n
        #pragma unroll
        for (int r = 0; r < 8; ++r) {
            const int m = m0 + r + laneH * 8;
            const float4 u = inp4[(long)m * TT + t];
            const float d = u.x * iw0 + u.y * iw1 + u.z * iw2 + u.w * iw3;
            const float hold = hc32[(long)m * H3 + n];
            float v = 0.9f * hold + 0.1f * (acc[r] + d);
            v = fmaxf(v, 0.0f);
            hx32[(long)m * H3 + n] = v;
            hx16[(long)m * H3 + n] = (_Float16)v;
            out_rnn[((long)m * TT + t) * H3 + n] = v;
            if (t == TT - 1) out_hn[(long)m * H3 + n] = v;
        }

        grid_barrier(sync, sync + 1, NBLOCKS_SCAN);
    }
}

// ---- output heads: mean/std over ALM region (n >= 2048), A == 1 ------------
__global__ void head_kernel(const float* __restrict__ rnn_out, // [32,512,3072]
                            const float* __restrict__ mean_w,  // [1,3072]
                            const float* __restrict__ mean_b,  // [1]
                            const float* __restrict__ std_w,
                            const float* __restrict__ std_b,
                            float* __restrict__ mean_out,      // [32,512]
                            float* __restrict__ std_out) {
    const int wv   = threadIdx.x >> 5;
    const int lane = threadIdx.x & 31;
    const long row = (long)blockIdx.x * 8 + wv;    // b*T + t
    if (row >= (long)BB * TT) return;
    const float4* x  = reinterpret_cast<const float4*>(rnn_out + row * H3 + 2 * HH);
    const float4* mw = reinterpret_cast<const float4*>(mean_w + 2 * HH);
    const float4* sw = reinterpret_cast<const float4*>(std_w + 2 * HH);
    float sm = 0.0f, ss = 0.0f;
    #pragma unroll 2
    for (int j = lane; j < HH / 4; j += 32) {
        const float4 v = x[j];
        const float4 wm = mw[j];
        const float4 ws = sw[j];
        sm += v.x * wm.x + v.y * wm.y + v.z * wm.z + v.w * wm.w;
        ss += v.x * ws.x + v.y * ws.y + v.z * ws.z + v.w * ws.w;
    }
    #pragma unroll
    for (int o = 16; o; o >>= 1) {
        sm += __shfl_xor(sm, o, 32);
        ss += __shfl_xor(ss, o, 32);
    }
    if (lane == 0) {
        mean_out[row] = sm + mean_b[0];
        std_out[row]  = fminf(fmaxf(ss + std_b[0], -5.0f), 10.0f);
    }
}

// ---------------------------------------------------------------------------
extern "C" void kernel_launch(void* const* d_in, const int* in_sizes, int n_in,
                              void* d_out, int out_size, void* d_ws, size_t ws_size,
                              hipStream_t stream) {
    const float* inp         = (const float*)d_in[0];
    const float* hn          = (const float*)d_in[1];
    // d_in[2] = w_str2str (unused: its mask is all-zero in the reference)
    const float* w_str2thal  = (const float*)d_in[3];
    const float* w_m12m1     = (const float*)d_in[4];
    const float* w_m12str    = (const float*)d_in[5];
    const float* w_thal2m1   = (const float*)d_in[6];
    const float* fixedw      = (const float*)d_in[7];
    const float* inpw        = (const float*)d_in[8];
    const float* mean_w      = (const float*)d_in[9];
    const float* mean_b      = (const float*)d_in[10];
    const float* std_w       = (const float*)d_in[11];
    const float* std_b       = (const float*)d_in[12];
    float* out = (float*)d_out;

    // workspace layout
    char* ws = (char*)d_ws;
    _Float16* W16 = (_Float16*)(ws);                 // 18,874,368 B
    float*    h32 = (float*)(ws + 18874368);         // 2 x 393,216 B (ping-pong)
    _Float16* h16 = (_Float16*)(ws + 19660800);      // 2 x 196,608 B (ping-pong)
    unsigned* sync = (unsigned*)(ws + 20054016);     //           8 B

    // 1) init h state + sync counters
    init_state_kernel<<<(BB * H3 + 255) / 256, 256, 0, stream>>>(hn, h32, h16, sync);
    // 2) build f16 W_rec
    build_wrec_kernel<<<(int)(((long)H3 * H3 + 255) / 256), 256, 0, stream>>>(
        w_str2thal, w_m12m1, w_m12str, w_thal2m1, fixedw, W16);
    // 3) persistent sequential scan (512 steps, grid barrier per step)
    rnn_scan_kernel<<<NBLOCKS_SCAN, 128, 0, stream>>>(
        W16, inp, inpw, h32, h16, out, sync);
    // 4) mean/std heads from rnn_out
    head_kernel<<<(BB * TT) / 8, 256, 0, stream>>>(
        out + OUT_RNN_OFF, mean_w, mean_b, std_w, std_b,
        out + OUT_MEAN_OFF, out + OUT_STD_OFF);
}
